// DataDepHebbian_10428180594724
// MI455X (gfx1250) — matrix operand order repaired and checked by reading, not compiled, not run on hardware
//
#include <hip/hip_runtime.h>

// ---------------------------------------------------------------------------
// Problem constants (from reference): B=4, T=2048, D=1024, M=256
// ---------------------------------------------------------------------------
static constexpr int Bn = 4, Tn = 2048, Dn = 1024, Mn = 256;
static constexpr int RN = Bn * Tn;  // 8192 flattened rows

typedef __attribute__((ext_vector_type(16))) __bf16 v16bf;
typedef __attribute__((ext_vector_type(8)))  __bf16 v8bf;
typedef __attribute__((ext_vector_type(8)))  float  v8f;

#define DEV __device__ __forceinline__

DEV v16bf cat8(v8bf lo, v8bf hi) {
  return __builtin_shufflevector(lo, hi, 0,1,2,3,4,5,6,7,8,9,10,11,12,13,14,15);
}

// A fragment (16x32 bf16), row-major source, contraction along fastest dim.
// ISA layout: lanes 0-15 = rows, K={0..7}+{16..23}; lanes 16-31: K={8..15}+{24..31}.
DEV v16bf load_a_frag(const __bf16* base, int lda, int r0, int k0, int lane) {
  int row = r0 + (lane & 15);
  int ko  = (lane < 16) ? 0 : 8;
  const __bf16* p = base + (size_t)row * lda + k0 + ko;
  v8bf lo = *(const v8bf*)p;
  v8bf hi = *(const v8bf*)(p + 16);
  return cat8(lo, hi);
}

// B fragment (32x16 bf16) built from W[N][K] row-major: B[k][n] = W[n0+n][k0+k].
// ISA layout: lanes 0-15 = cols n, K={0..7}+{16..23}; lanes 16-31: K={8..15}+{24..31}.
DEV v16bf load_b_frag(const __bf16* base, int ldk, int n0, int k0, int lane) {
  int col = n0 + (lane & 15);
  int ko  = (lane < 16) ? 0 : 8;
  const __bf16* p = base + (size_t)col * ldk + k0 + ko;
  v8bf lo = *(const v8bf*)p;
  v8bf hi = *(const v8bf*)(p + 16);
  return cat8(lo, hi);
}

DEV v8f wmma_bf16(v16bf a, v16bf b, v8f c) {
  // (neg_a, A, neg_b, B, c_mod, C, reuse_a, reuse_b)
  return __builtin_amdgcn_wmma_f32_16x16x32_bf16(false, a, false, b, (short)0, c,
                                                 false, false);
}

// ---------------------------------------------------------------------------
// fp32 -> bf16 conversion (grid-stride)
// ---------------------------------------------------------------------------
__global__ void cvt_bf16(const float* __restrict__ in, __bf16* __restrict__ out, int n) {
  for (int i = blockIdx.x * blockDim.x + threadIdx.x; i < n; i += gridDim.x * blockDim.x)
    out[i] = (__bf16)in[i];
}

// ---------------------------------------------------------------------------
// log(sigmoid(x . Wg + bg) + 1e-8) per (b,t) row, fp32 (numerically sensitive)
// ---------------------------------------------------------------------------
__global__ __launch_bounds__(256) void gate_kernel(const float* __restrict__ x,
                                                   const float* __restrict__ Wg,
                                                   const float* __restrict__ bg,
                                                   float* __restrict__ lg) {
  __shared__ float red[256];
  const int row = blockIdx.x, tid = threadIdx.x;
  const float* xr = x + (size_t)row * Dn;
  float p = 0.f;
  for (int k = tid; k < Dn; k += 256) p += xr[k] * Wg[k];
  red[tid] = p;
  __syncthreads();
  for (int s = 128; s > 0; s >>= 1) {
    if (tid < s) red[tid] += red[tid + s];
    __syncthreads();
  }
  if (tid == 0) {
    float z = red[0] + bg[0];
    float g = 1.f / (1.f + __expf(-z));
    lg[row] = __logf(g + 1e-8f);
  }
}

// ---------------------------------------------------------------------------
// In-place inclusive cumsum over T per batch (1 block per batch)
// ---------------------------------------------------------------------------
__global__ __launch_bounds__(256) void cumsum_kernel(float* __restrict__ cl) {
  __shared__ float s[256];
  const int b = blockIdx.x, tid = threadIdx.x;
  float carry = 0.f;
  for (int c0 = 0; c0 < Tn; c0 += 256) {
    float v = cl[b * Tn + c0 + tid];
    s[tid] = v;
    __syncthreads();
    for (int off = 1; off < 256; off <<= 1) {
      float t = (tid >= off) ? s[tid - off] : 0.f;
      __syncthreads();
      s[tid] += t;
      __syncthreads();
    }
    cl[b * Tn + c0 + tid] = s[tid] + carry;
    float tot = s[255];
    __syncthreads();
    carry += tot;
  }
}

// ---------------------------------------------------------------------------
// Register-blocked bf16 GEMM: out[r][n] = sum_k A[r][k] * W[n][k]
// grid = (R/64, N/128), block = 128 (4 waves).
// Wave computes a 32x64 macro-tile = 2x4 WMMA tiles; per k-step:
//   4 A-loads + 8 B-loads (b128) feed 8 WMMAs -> 1.5 loads/WMMA.
// ---------------------------------------------------------------------------
__global__ __launch_bounds__(128) void gemm_bf16(const __bf16* __restrict__ A,
                                                 const __bf16* __restrict__ W,
                                                 float* __restrict__ outF,
                                                 __bf16* __restrict__ outH,
                                                 int R, int N, int K) {
  const int lane = threadIdx.x & 31, wave = threadIdx.x >> 5;
  const int r0 = blockIdx.x * 64 + (wave & 1) * 32;
  const int n0 = blockIdx.y * 128 + (wave >> 1) * 64;
  (void)R;

  const v8f vzero = {0.f, 0.f, 0.f, 0.f, 0.f, 0.f, 0.f, 0.f};
  v8f acc[2][4];
#pragma unroll
  for (int i = 0; i < 2; ++i)
#pragma unroll
    for (int c = 0; c < 4; ++c) acc[i][c] = vzero;

  for (int k0 = 0; k0 < K; k0 += 32) {
    // prefetch the K-stream ~4 iterations ahead (speculative; OOB is dropped)
    __builtin_prefetch((const void*)(A + (size_t)(r0 + (lane & 15)) * K + k0 + 128), 0, 3);
    __builtin_prefetch((const void*)(W + (size_t)(n0 + (lane & 15)) * K + k0 + 128), 0, 3);

    v16bf a0 = load_a_frag(A, K, r0, k0, lane);
    v16bf a1 = load_a_frag(A, K, r0 + 16, k0, lane);
    v16bf b0 = load_b_frag(W, K, n0, k0, lane);
    v16bf b1 = load_b_frag(W, K, n0 + 16, k0, lane);
    v16bf b2 = load_b_frag(W, K, n0 + 32, k0, lane);
    v16bf b3 = load_b_frag(W, K, n0 + 48, k0, lane);

    acc[0][0] = wmma_bf16(a0, b0, acc[0][0]);
    acc[0][1] = wmma_bf16(a0, b1, acc[0][1]);
    acc[0][2] = wmma_bf16(a0, b2, acc[0][2]);
    acc[0][3] = wmma_bf16(a0, b3, acc[0][3]);
    acc[1][0] = wmma_bf16(a1, b0, acc[1][0]);
    acc[1][1] = wmma_bf16(a1, b1, acc[1][1]);
    acc[1][2] = wmma_bf16(a1, b2, acc[1][2]);
    acc[1][3] = wmma_bf16(a1, b3, acc[1][3]);
  }

  // C layout: VGPR g holds row g (lanes 0-15) / row g+8 (lanes 16-31), col = lane%16
  const int colb  = lane & 15;
  const int rhalf = (lane < 16) ? 0 : 8;
#pragma unroll
  for (int i = 0; i < 2; ++i) {
#pragma unroll
    for (int c = 0; c < 4; ++c) {
      const int col   = n0 + c * 16 + colb;
      const int rbase = r0 + i * 16 + rhalf;
      if (outF) {
#pragma unroll
        for (int g = 0; g < 8; ++g) outF[(size_t)(rbase + g) * N + col] = acc[i][c][g];
      } else {
#pragma unroll
        for (int g = 0; g < 8; ++g) outH[(size_t)(rbase + g) * N + col] = (__bf16)acc[i][c][g];
      }
    }
  }
}

// ---------------------------------------------------------------------------
// Fused gated attention:
//  reads[b,j,m] = (1/sqrt(T)) * sum_{i<=j} (V[i].Q[j]/sqrt(M)) *
//                 exp(min(clog[j]-clog[i],0)) * K[i,m]
// Block: 256 threads (8 waves); 64 j's x full M=256 per block.
// Phase A: 64x64 score tile via WMMA (C rows=i, cols=j), decay+mask, store
//          transposed bf16 into LDS wT[j][i].
// Phase B: acc[j][m] += wT(64x64) x K_i(64x256); K_i staged transposed in LDS
//          KT[m][i] so B fragments are contiguous along the contraction (i).
// ---------------------------------------------------------------------------
__global__ __launch_bounds__(256) void attn_kernel(const __bf16* __restrict__ Kb,
                                                   const __bf16* __restrict__ Vb,
                                                   const __bf16* __restrict__ Qb,
                                                   const float* __restrict__ clog,
                                                   __bf16* __restrict__ Rb) {
  __shared__ __bf16 wT[64][72];   // stride 72: rows 16B aligned, odd 16B-unit lane stride
  __shared__ __bf16 KT[256][72];

  const int tid = threadIdx.x, lane = tid & 31, wave = tid >> 5;
  const int b = blockIdx.y, j0 = blockIdx.x * 64;
  const float inv_sqrtM = 0.0625f;               // 1/sqrt(256)
  const float inv_sqrtT = 0.02209708691207961f;  // 1/sqrt(2048)

  const __bf16* Kbase = Kb + (size_t)b * Tn * Mn;
  const __bf16* Vbase = Vb + (size_t)b * Tn * Mn;
  const __bf16* Qbase = Qb + (size_t)b * Tn * Mn;
  const float*  cl    = clog + b * Tn;

  // phase A tile assignment: wave -> (ii = wave&3, jj in {wave>>2, (wave>>2)+2})
  const int iiA  = wave & 3;
  const int jjA  = wave >> 2;
  // phase B assignment: wave -> j-subtile (wave&3), m-half (wave>>2)
  const int jsub  = wave & 3;
  const int mhalf = wave >> 2;

  v8f acc[8];
  const v8f vzero = {0.f, 0.f, 0.f, 0.f, 0.f, 0.f, 0.f, 0.f};
#pragma unroll
  for (int t = 0; t < 8; ++t) acc[t] = vzero;

  const int ntiles = j0 / 64 + 1;  // causal: i-tiles with i0 <= j0
  for (int it = 0; it < ntiles; ++it) {
    const int i0 = it * 64;
    __syncthreads();  // previous phase B finished reading wT/KT

    // --- stage KT[m][i'] (transpose of K tile). Thread tid owns column m=tid.
    {
      const __bf16* kp = Kbase + (size_t)i0 * Mn + tid;
      // prefetch next i-tile's K column while we consume this one
      __builtin_prefetch((const void*)(kp + 64 * Mn), 0, 3);
      __builtin_prefetch((const void*)(Vbase + (size_t)(i0 + 64) * Mn + tid), 0, 3);
#pragma unroll
      for (int ic = 0; ic < 8; ++ic) {
        v8bf v;
#pragma unroll
        for (int e = 0; e < 8; ++e) v[e] = kp[(size_t)(ic * 8 + e) * Mn];
        *(v8bf*)&KT[tid][ic * 8] = v;  // 16B-aligned ds store
      }
    }

    // --- phase A: two 16x16 score tiles per wave, K-loop over M
    {
      v8f s0 = vzero, s1 = vzero;
      for (int k0 = 0; k0 < Mn; k0 += 32) {
        v16bf av  = load_a_frag(Vbase + (size_t)i0 * Mn, Mn, iiA * 16, k0, lane);
        v16bf bq0 = load_b_frag(Qbase + (size_t)j0 * Mn, Mn, jjA * 16, k0, lane);
        v16bf bq1 = load_b_frag(Qbase + (size_t)j0 * Mn, Mn, (jjA + 2) * 16, k0, lane);
        s0 = wmma_bf16(av, bq0, s0);
        s1 = wmma_bf16(av, bq1, s1);
      }
      // decay * causal mask, scale by 1/sqrt(M), store transposed into wT
      const int jc0 = jjA * 16 + (lane & 15);
      const int jc1 = (jjA + 2) * 16 + (lane & 15);
      const int ib  = iiA * 16 + ((lane < 16) ? 0 : 8);
      const float cj0 = cl[j0 + jc0];
      const float cj1 = cl[j0 + jc1];
#pragma unroll
      for (int g = 0; g < 8; ++g) {
        const int irow = ib + g;
        const int gi   = i0 + irow;
        const float ci = cl[gi];  // broadcast load within half-wave
        const float d0 = (gi <= j0 + jc0) ? __expf(fminf(cj0 - ci, 0.f)) : 0.f;
        const float d1 = (gi <= j0 + jc1) ? __expf(fminf(cj1 - ci, 0.f)) : 0.f;
        wT[jc0][irow] = (__bf16)(s0[g] * inv_sqrtM * d0);
        wT[jc1][irow] = (__bf16)(s1[g] * inv_sqrtM * d1);
      }
    }
    __syncthreads();  // wT + KT ready

    // --- phase B: acc[j][m] += wT[j][i'] x KT[m][i'] (contraction over i' = 64)
#pragma unroll
    for (int ks = 0; ks < 2; ++ks) {
      const int k0 = ks * 32;
      v16bf aw = load_a_frag(&wT[0][0], 72, jsub * 16, k0, lane);
#pragma unroll
      for (int ms = 0; ms < 8; ++ms) {
        v16bf bk = load_b_frag(&KT[0][0], 72, mhalf * 128 + ms * 16, k0, lane);
        acc[ms] = wmma_bf16(aw, bk, acc[ms]);
      }
    }
  }

  // epilogue: reads * (1/sqrt(T)) -> bf16 for the output projection
  {
    const int col = lane & 15;
    const int jb  = jsub * 16 + ((lane < 16) ? 0 : 8);
#pragma unroll
    for (int ms = 0; ms < 8; ++ms) {
      const int m = mhalf * 128 + ms * 16 + col;
#pragma unroll
      for (int g = 0; g < 8; ++g) {
        Rb[((size_t)b * Tn + j0 + jb + g) * Mn + m] = (__bf16)(acc[ms][g] * inv_sqrtT);
      }
    }
  }
}

// ---------------------------------------------------------------------------
// Host launch
// ---------------------------------------------------------------------------
extern "C" void kernel_launch(void* const* d_in, const int* in_sizes, int n_in,
                              void* d_out, int out_size, void* d_ws, size_t ws_size,
                              hipStream_t stream) {
  (void)in_sizes; (void)n_in; (void)out_size; (void)ws_size;
  const float* x  = (const float*)d_in[0];
  const float* Wk = (const float*)d_in[1];
  const float* Wv = (const float*)d_in[2];
  const float* Wq = (const float*)d_in[3];
  const float* Wg = (const float*)d_in[4];
  const float* bg = (const float*)d_in[5];
  const float* Wo = (const float*)d_in[6];
  float* out = (float*)d_out;

  char* w = (char*)d_ws;
  auto carve = [&](size_t bytes) -> char* {
    char* p = w;
    w += (bytes + 255) & ~(size_t)255;
    return p;
  };
  __bf16* xh   = (__bf16*)carve((size_t)RN * Dn * 2);  // 16 MB
  __bf16* Wkh  = (__bf16*)carve((size_t)Mn * Dn * 2);
  __bf16* Wvh  = (__bf16*)carve((size_t)Mn * Dn * 2);
  __bf16* Wqh  = (__bf16*)carve((size_t)Mn * Dn * 2);
  __bf16* Woh  = (__bf16*)carve((size_t)Dn * Mn * 2);
  __bf16* Kbuf = (__bf16*)carve((size_t)RN * Mn * 2);  // 4 MB each
  __bf16* Vbuf = (__bf16*)carve((size_t)RN * Mn * 2);
  __bf16* Qbuf = (__bf16*)carve((size_t)RN * Mn * 2);
  __bf16* Rbuf = (__bf16*)carve((size_t)RN * Mn * 2);
  float*  clb  = (float*)carve((size_t)RN * 4);        // cum_log (in-place scan)

  // 1) precision conversion
  cvt_bf16<<<2048, 256, 0, stream>>>(x, xh, RN * Dn);
  cvt_bf16<<<256, 256, 0, stream>>>(Wk, Wkh, Mn * Dn);
  cvt_bf16<<<256, 256, 0, stream>>>(Wv, Wvh, Mn * Dn);
  cvt_bf16<<<256, 256, 0, stream>>>(Wq, Wqh, Mn * Dn);
  cvt_bf16<<<256, 256, 0, stream>>>(Wo, Woh, Dn * Mn);

  // 2) gates (fp32) + cumulative log-gate scan
  gate_kernel<<<RN, 256, 0, stream>>>(x, Wg, bg, clb);
  cumsum_kernel<<<Bn, 256, 0, stream>>>(clb);

  // 3) K/V/Q projections (bf16 WMMA, bf16 outputs)
  dim3 gproj(RN / 64, Mn / 128);
  gemm_bf16<<<gproj, 128, 0, stream>>>(xh, Wkh, nullptr, Kbuf, RN, Mn, Dn);
  gemm_bf16<<<gproj, 128, 0, stream>>>(xh, Wvh, nullptr, Vbuf, RN, Mn, Dn);
  gemm_bf16<<<gproj, 128, 0, stream>>>(xh, Wqh, nullptr, Qbuf, RN, Mn, Dn);

  // 4) fused causal decayed attention
  attn_kernel<<<dim3(Tn / 64, Bn), 256, 0, stream>>>(Kbuf, Vbuf, Qbuf, clb, Rbuf);

  // 5) output projection (fp32 result)
  gemm_bf16<<<dim3(RN / 64, Dn / 128), 128, 0, stream>>>(Rbuf, Woh, out, nullptr,
                                                         RN, Dn, Mn);
}
